// GCN_23725399343418
// MI455X (gfx1250) — matrix operand order repaired and checked by reading, not compiled, not run on hardware
//
#include <hip/hip_runtime.h>

typedef __attribute__((ext_vector_type(16))) _Float16 v16h;
typedef __attribute__((ext_vector_type(8)))  _Float16 v8h;
typedef __attribute__((ext_vector_type(8)))  float    v8f;

#define N_NODES 100000
#define N_EDGES 1600000
#define D_IN    256
#define D_HID   256
#define D_OUT   64

#define WMMA_F16(a, b, c) \
  __builtin_amdgcn_wmma_f32_16x16x32_f16(false, (a), false, (b), (short)0, (c), false, false)

// ---------------------------------------------------------------------------
// Zero-fill (workspace/d_out are poisoned; must re-zero every call)
// ---------------------------------------------------------------------------
__global__ __launch_bounds__(256) void k_zero(float* p, int n) {
  int i = blockIdx.x * 256 + threadIdx.x;
  if (i < n) p[i] = 0.0f;
}

// ---------------------------------------------------------------------------
// Convert + transpose weights to f16 so WMMA B-fragments are contiguous:
//   w1t[n][k] = f16(W1[k][n])  (256x256),  w2t[n][k] = f16(W2[k][n]) (64x256)
// ---------------------------------------------------------------------------
__global__ __launch_bounds__(256) void k_convert_w(const float* __restrict__ W1,
                                                   const float* __restrict__ W2,
                                                   _Float16* __restrict__ w1t,
                                                   _Float16* __restrict__ w2t) {
  int t = blockIdx.x * 256 + threadIdx.x;
  if (t < 256 * 256) {
    int k = t >> 8, n = t & 255;
    w1t[n * 256 + k] = (_Float16)W1[k * 256 + n];
  } else {
    t -= 256 * 256;
    if (t < 256 * 64) {
      int k = t >> 6, n = t & 63;
      w2t[n * 256 + k] = (_Float16)W2[k * 64 + n];
    }
  }
}

// ---------------------------------------------------------------------------
// GEMM1: H1(f16) = x(f32, N x 256) @ W1 (via w1t f16).
// One wave = 16 rows x 64 cols (4 accumulators): each K-step loads one
// A-fragment (converted f32->f16 once) and 4 B-fragments -> 4 WMMAs.
// Fragment layouts used here (contiguous k order, same in A and B):
//   A (16x32 f16): lane (h=lane>>4, m=lane&15); elem i -> K=k0+16h+i
//                  (16 consecutive f32 of x row m, four float4 loads)
//   B (32x16 f16): lane (h, n=lane&15); elem i -> K=k0+16h+i (32B of w*t row n)
//   C/D (16x16 f32): lane (h, n); vgpr v -> row 8h+v
// ---------------------------------------------------------------------------
__global__ __launch_bounds__(256) void k_gemm_xw1(const float* __restrict__ x,
                                                  const _Float16* __restrict__ w1t,
                                                  _Float16* __restrict__ h1) {
  const int lane = threadIdx.x & 31;
  const int wave = threadIdx.x >> 5;
  const int wt   = blockIdx.x * 8 + wave;     // 6250*4 wave-tiles (exact)
  const int rowTile = wt >> 2;                // 0..6249
  const int colGrp  = wt & 3;                 // 0..3, 64 cols each
  const int hh  = lane >> 4;
  const int l16 = lane & 15;
  const float*    arow = x + (size_t)(rowTile * 16 + l16) * 256;
  const _Float16* bp0 = w1t + (size_t)(colGrp * 64 +  0 + l16) * 256;
  const _Float16* bp1 = w1t + (size_t)(colGrp * 64 + 16 + l16) * 256;
  const _Float16* bp2 = w1t + (size_t)(colGrp * 64 + 32 + l16) * 256;
  const _Float16* bp3 = w1t + (size_t)(colGrp * 64 + 48 + l16) * 256;
  v8f acc0 = {}, acc1 = {}, acc2 = {}, acc3 = {};
#pragma unroll
  for (int k0 = 0; k0 < 256; k0 += 32) {
    v16h a;
    const float4 u0 = *(const float4*)(arow + k0 + 16 * hh);
    const float4 u1 = *(const float4*)(arow + k0 + 16 * hh + 4);
    const float4 u2 = *(const float4*)(arow + k0 + 16 * hh + 8);
    const float4 u3 = *(const float4*)(arow + k0 + 16 * hh + 12);
    a[0]  = (_Float16)u0.x; a[1]  = (_Float16)u0.y; a[2]  = (_Float16)u0.z; a[3]  = (_Float16)u0.w;
    a[4]  = (_Float16)u1.x; a[5]  = (_Float16)u1.y; a[6]  = (_Float16)u1.z; a[7]  = (_Float16)u1.w;
    a[8]  = (_Float16)u2.x; a[9]  = (_Float16)u2.y; a[10] = (_Float16)u2.z; a[11] = (_Float16)u2.w;
    a[12] = (_Float16)u3.x; a[13] = (_Float16)u3.y; a[14] = (_Float16)u3.z; a[15] = (_Float16)u3.w;
    const int ko = k0 + 16 * hh;
    v16h b0 = *(const v16h*)(bp0 + ko);
    v16h b1 = *(const v16h*)(bp1 + ko);
    v16h b2 = *(const v16h*)(bp2 + ko);
    v16h b3 = *(const v16h*)(bp3 + ko);
    acc0 = WMMA_F16(a, b0, acc0);
    acc1 = WMMA_F16(a, b1, acc1);
    acc2 = WMMA_F16(a, b2, acc2);
    acc3 = WMMA_F16(a, b3, acc3);
  }
  const int baseRow = rowTile * 16 + 8 * hh;
  const int baseCol = colGrp * 64 + l16;
#pragma unroll
  for (int vv = 0; vv < 8; ++vv) {
    _Float16* orow = h1 + (size_t)(baseRow + vv) * 256 + baseCol;
    orow[0]  = (_Float16)acc0[vv];
    orow[16] = (_Float16)acc1[vv];
    orow[32] = (_Float16)acc2[vv];
    orow[48] = (_Float16)acc3[vv];
  }
}

// ---------------------------------------------------------------------------
// GEMM2: H3(f16, N x 64) = h(f16, N x 256) @ W2 (via w2t f16).
// One wave covers the whole 64-col output stripe (4 accumulators).
// 6250 waves -> 782 blocks; wave-uniform guard keeps EXEC all-ones.
// ---------------------------------------------------------------------------
__global__ __launch_bounds__(256) void k_gemm_hw2(const _Float16* __restrict__ h2,
                                                  const _Float16* __restrict__ w2t,
                                                  _Float16* __restrict__ h3) {
  const int lane = threadIdx.x & 31;
  const int wave = threadIdx.x >> 5;
  const int rowTile = blockIdx.x * 8 + wave;  // 0..6249 (+ tail waves)
  if (rowTile >= N_NODES / 16) return;        // wave-uniform branch
  const int hh  = lane >> 4;
  const int l16 = lane & 15;
  const _Float16* arow = h2 + (size_t)(rowTile * 16 + l16) * 256;
  const _Float16* bp0 = w2t + (size_t)( 0 + l16) * 256;
  const _Float16* bp1 = w2t + (size_t)(16 + l16) * 256;
  const _Float16* bp2 = w2t + (size_t)(32 + l16) * 256;
  const _Float16* bp3 = w2t + (size_t)(48 + l16) * 256;
  v8f acc0 = {}, acc1 = {}, acc2 = {}, acc3 = {};
#pragma unroll
  for (int k0 = 0; k0 < 256; k0 += 32) {
    v16h a;
    ((v8h*)&a)[0] = *(const v8h*)(arow + k0 + 16 * hh);
    ((v8h*)&a)[1] = *(const v8h*)(arow + k0 + 16 * hh + 8);
    const int ko = k0 + 16 * hh;
    v16h b0 = *(const v16h*)(bp0 + ko);
    v16h b1 = *(const v16h*)(bp1 + ko);
    v16h b2 = *(const v16h*)(bp2 + ko);
    v16h b3 = *(const v16h*)(bp3 + ko);
    acc0 = WMMA_F16(a, b0, acc0);
    acc1 = WMMA_F16(a, b1, acc1);
    acc2 = WMMA_F16(a, b2, acc2);
    acc3 = WMMA_F16(a, b3, acc3);
  }
  const int baseRow = rowTile * 16 + 8 * hh;
#pragma unroll
  for (int vv = 0; vv < 8; ++vv) {
    _Float16* orow = h3 + (size_t)(baseRow + vv) * 64 + l16;
    orow[0]  = (_Float16)acc0[vv];
    orow[16] = (_Float16)acc1[vv];
    orow[32] = (_Float16)acc2[vv];
    orow[48] = (_Float16)acc3[vv];
  }
}

__device__ __forceinline__ void fadd_atomic(float* p, float v) {
  __hip_atomic_fetch_add(p, v, __ATOMIC_RELAXED, __HIP_MEMORY_SCOPE_AGENT);
}

// ---------------------------------------------------------------------------
// COO SpMM scatter, hidden layer (256 feats): thread = (edge, 8-feat chunk).
// Gather h[col] rows from f16 (51.2 MB -> resident in 192 MB L2), scatter
// val*h into fp32 accumulator with native global_atomic_add_f32.
// ---------------------------------------------------------------------------
__global__ __launch_bounds__(256) void k_spmm_hid(const int* __restrict__ row,
                                                  const int* __restrict__ col,
                                                  const float* __restrict__ val,
                                                  const _Float16* __restrict__ h,
                                                  float* __restrict__ acc) {
  long tid = (long)blockIdx.x * 256 + threadIdx.x;     // E * 32 threads (exact)
  int e  = (int)(tid >> 5);
  int d0 = ((int)tid & 31) * 8;
  int r = row[e], c = col[e];
  float v = val[e];
  v8h hv = *(const v8h*)(h + (size_t)c * 256 + d0);
  float* dst = acc + (size_t)r * 256 + d0;
#pragma unroll
  for (int i = 0; i < 8; ++i) fadd_atomic(dst + i, v * (float)hv[i]);
}

// SpMM scatter, output layer (64 feats): thread = (edge, 8-feat chunk)
__global__ __launch_bounds__(256) void k_spmm_out(const int* __restrict__ row,
                                                  const int* __restrict__ col,
                                                  const float* __restrict__ val,
                                                  const _Float16* __restrict__ h,
                                                  float* __restrict__ out) {
  long tid = (long)blockIdx.x * 256 + threadIdx.x;     // E * 8 threads (exact)
  int e  = (int)(tid >> 3);
  int d0 = ((int)tid & 7) * 8;
  int r = row[e], c = col[e];
  float v = val[e];
  v8h hv = *(const v8h*)(h + (size_t)c * 64 + d0);
  float* dst = out + (size_t)r * 64 + d0;
#pragma unroll
  for (int i = 0; i < 8; ++i) fadd_atomic(dst + i, v * (float)hv[i]);
}

// h2 = f16(relu(acc + b1)), written over the (now free) H1 region
__global__ __launch_bounds__(256) void k_bias_relu(const float* __restrict__ acc,
                                                   const float* __restrict__ b1,
                                                   _Float16* __restrict__ h2) {
  int i = blockIdx.x * 256 + threadIdx.x;   // N*256 exact
  float t = acc[i] + b1[i & 255];
  h2[i] = (_Float16)(t > 0.0f ? t : 0.0f);
}

// out += b2
__global__ __launch_bounds__(256) void k_bias_out(float* __restrict__ out,
                                                  const float* __restrict__ b2) {
  int i = blockIdx.x * 256 + threadIdx.x;   // N*64 exact
  out[i] += b2[i & 63];
}

// ---------------------------------------------------------------------------
// Workspace layout (bytes), total ~166.6 MB:
//   [0,       131072)             w1t  f16 (W1^T)
//   [131072,  163840)             w2t  f16 (W2^T)
//   [163840,  +N*256*2)           h16  f16 (H1, later reused as H2)
//   [...,     +N*256*4)           acc  f32 scatter accumulator
//   [...,     +N*64*2)            h3   f16 (H3)
// ---------------------------------------------------------------------------
extern "C" void kernel_launch(void* const* d_in, const int* in_sizes, int n_in,
                              void* d_out, int out_size, void* d_ws, size_t ws_size,
                              hipStream_t stream) {
  (void)in_sizes; (void)n_in; (void)out_size; (void)ws_size;
  const float* x    = (const float*)d_in[0];
  const int*   arow = (const int*)  d_in[1];
  const int*   acol = (const int*)  d_in[2];
  const float* aval = (const float*)d_in[3];
  const float* W1   = (const float*)d_in[4];
  const float* b1   = (const float*)d_in[5];
  const float* W2   = (const float*)d_in[6];
  const float* b2   = (const float*)d_in[7];
  float* out = (float*)d_out;

  char* ws = (char*)d_ws;
  _Float16* w1t = (_Float16*)(ws);
  _Float16* w2t = (_Float16*)(ws + 131072);
  _Float16* h16 = (_Float16*)(ws + 163840);
  float*    acc = (float*)   (ws + 163840 + (size_t)N_NODES * 256 * 2);
  _Float16* h3  = (_Float16*)(ws + 163840 + (size_t)N_NODES * 256 * 2
                                          + (size_t)N_NODES * 256 * 4);

  // weights: convert + transpose (tiny)
  k_convert_w<<<320, 256, 0, stream>>>(W1, W2, w1t, w2t);
  // zero accumulators
  k_zero<<<(N_NODES * 256) / 256, 256, 0, stream>>>(acc, N_NODES * 256);
  k_zero<<<(N_NODES * 64) / 256, 256, 0, stream>>>(out, N_NODES * 64);
  // layer 1
  k_gemm_xw1<<<(N_NODES / 16) * 4 / 8, 256, 0, stream>>>(x, w1t, h16);        // 3125 blocks
  k_spmm_hid<<<(int)(((long)N_EDGES * 32) / 256), 256, 0, stream>>>(arow, acol, aval, h16, acc);
  k_bias_relu<<<(N_NODES * 256) / 256, 256, 0, stream>>>(acc, b1, h16);
  // layer 2
  k_gemm_hw2<<<(N_NODES / 16 + 7) / 8, 256, 0, stream>>>(h16, w2t, h3);       // 782 blocks
  k_spmm_out<<<(int)(((long)N_EDGES * 8) / 256), 256, 0, stream>>>(arow, acol, aval, h3, out);
  k_bias_out<<<(N_NODES * 64) / 256, 256, 0, stream>>>(out, b2);
}
